// CedricsGATLayer_63771674411015
// MI455X (gfx1250) — matrix operand order repaired and checked by reading, not compiled, not run on hardware
//
#include <hip/hip_runtime.h>
#include <hip/hip_bf16.h>

typedef __attribute__((ext_vector_type(16))) _Float16 v16h;
typedef __attribute__((ext_vector_type(4)))  _Float16 v4h;
typedef __attribute__((ext_vector_type(8)))  float    v8f;
typedef __attribute__((ext_vector_type(4)))  float    v4f;

// sizes
#define BB 8
#define NN 512

// ---------------------------------------------------------------------------
// k0: zero the output (relu outputs are >= 0, so 0 is the max-identity)
// ---------------------------------------------------------------------------
__global__ void k0_zero(float* out, int n) {
  int i = blockIdx.x * 256 + threadIdx.x;
  if (i < n) out[i] = 0.0f;
}

// ---------------------------------------------------------------------------
// k1: per-node precompute.  One 64-thread block per (b,n) row.
//   t0 = relu(coords@fw0+fb0)           (64)
//   h  = relu(t0@fw1+fb1)               (128)  -> h32 and h16
//   u  = coords@aw0[0:3]  v = coords@aw0[3:6]  (64 each)
// ---------------------------------------------------------------------------
__global__ __launch_bounds__(64) void k1_node(
    const float* __restrict__ coords,
    const float* __restrict__ fw0, const float* __restrict__ fb0,
    const float* __restrict__ fw1, const float* __restrict__ fb1,
    const float* __restrict__ aw0,
    float* __restrict__ h32, _Float16* __restrict__ h16,
    float* __restrict__ u_g, float* __restrict__ v_g)
{
  int row = blockIdx.x;          // 0 .. B*N-1
  int t   = threadIdx.x;         // 0 .. 63
  __shared__ float t0[64];
  __shared__ float c3[3];
  if (t < 3) c3[t] = coords[row * 3 + t];
  __syncthreads();
  float x0 = c3[0], x1 = c3[1], x2 = c3[2];
  float a = fb0[t] + x0 * fw0[0 * 64 + t] + x1 * fw0[1 * 64 + t] + x2 * fw0[2 * 64 + t];
  t0[t] = a > 0.f ? a : 0.f;
  u_g[row * 64 + t] = x0 * aw0[0 * 64 + t] + x1 * aw0[1 * 64 + t] + x2 * aw0[2 * 64 + t];
  v_g[row * 64 + t] = x0 * aw0[3 * 64 + t] + x1 * aw0[4 * 64 + t] + x2 * aw0[5 * 64 + t];
  __syncthreads();
  for (int c = t; c < 128; c += 64) {
    float s = fb1[c];
#pragma unroll 8
    for (int k = 0; k < 64; ++k) s += t0[k] * fw1[k * 128 + c];
    s = s > 0.f ? s : 0.f;
    h32[row * 128 + c] = s;
    h16[row * 128 + c] = (_Float16)s;
  }
}

// ---------------------------------------------------------------------------
// k2: adjacency MLP (WMMA) + aggregation V = A @ h (WMMA).
// One 256-thread (8-wave) block per (b, 16-row i-tile).
// Fragment layouts used here (contiguous k, same for A and B):
//   A 16x32 f16 : lane = M + 16*(k/16),    elem = k%16
//                 (=> k..k+3 contiguous elems for k%4==0, same lane)
//   B 32x16 f16 : lane = n + 16*(k/16),    elem = k%16
//   C/D 16x16 f32: lane = n + 16*(M/8),    vgpr = M%8
// ---------------------------------------------------------------------------
__global__ __launch_bounds__(256) void k2_adj_agg(
    const float* __restrict__ u_g, const float* __restrict__ v_g,
    const _Float16* __restrict__ h16,
    const float* __restrict__ ab0, const float* __restrict__ aw1,
    const float* __restrict__ ab1, const float* __restrict__ aw2,
    const float* __restrict__ ab2, float* __restrict__ V_g)
{
  __shared__ alignas(16) float u_s[16 * 64];             // 4 KB  (includes ab0)
  __shared__ alignas(16) float v_s[16 * 64];             // 4 KB
  __shared__ alignas(32) _Float16 a1f[16 * 2 * 32 * 16]; // 32 KB  A-fragments, layer-1 act
  __shared__ alignas(32) _Float16 a2_s[256 * 64];        // 32 KB  layer-2 act, [pair][c]
  __shared__ alignas(32) _Float16 Af[32 * 16];           // 1 KB   A-matrix fragment (16i x 32j)
  __shared__ alignas(32) _Float16 hf[8 * 32 * 16];       // 8 KB   h-tile B-fragments
  __shared__ alignas(32) _Float16 w1f[4 * 2 * 32 * 16];  // 8 KB   aw1 B-fragments

  const int tid  = threadIdx.x;
  const int lane = tid & 31;
  const int wave = tid >> 5;
  const int b    = blockIdx.x >> 5;          // N/16 = 32 i-tiles per batch
  const int i0   = (blockIdx.x & 31) << 4;

  // --- one-time staging -----------------------------------------------------
  {
    int idx4 = tid * 4;                      // 1024 floats total
    v4f uu = *(const v4f*)&u_g[((size_t)b * NN + i0 + (idx4 >> 6)) * 64 + (idx4 & 63)];
#pragma unroll
    for (int s = 0; s < 4; ++s) uu[s] += ab0[(idx4 & 63) + s];
    *(v4f*)&u_s[idx4] = uu;
  }
  // aw1 -> B fragments: [ct][q][lane][elem]
  for (int s = 0; s < 16; ++s) {
    int e = tid * 16 + s;                    // 0..4095
    int ct = e >> 10, q = (e >> 9) & 1, ln = (e >> 4) & 31, el = e & 15;
    int n = ln & 15, kk = ((ln >> 4) << 4) + el, k = q * 32 + kk, c = ct * 16 + n;
    w1f[e] = (_Float16)aw1[k * 64 + c];
  }
  float bias1[4];
#pragma unroll
  for (int ct = 0; ct < 4; ++ct) bias1[ct] = ab1[ct * 16 + (lane & 15)];
  const float b2 = ab2[0];
  __syncthreads();

  // keep aw1 fragments in registers (4 col-tiles x 2 K-steps = 64 VGPRs)
  v16h w1r[4][2];
#pragma unroll
  for (int ct = 0; ct < 4; ++ct)
#pragma unroll
    for (int q = 0; q < 2; ++q)
      w1r[ct][q] = *(const v16h*)&w1f[((ct * 2 + q) * 32 + lane) * 16];

  v8f Vacc = {};   // V rows i0..i0+15, cols wave*16..wave*16+15

  for (int j0 = 0; j0 < NN; j0 += 32) {
    // stage h[j0:j0+32, :] into B fragments.  Each thread reads 16 consecutive
    // halfs of one h row (wide coalesced global load), scatters to fragments.
    {
      int e = tid * 16;
      int k = e >> 7, col0 = e & 127;        // col0 is 16-aligned
      v16h row = *(const v16h*)&h16[((size_t)b * NN + j0 + k) * 128 + col0];
      int ct = col0 >> 4;
      int kp = ((k >> 4) << 4);              // lane group offset
      int el = k & 15;
#pragma unroll
      for (int s = 0; s < 16; ++s)
        hf[ct * 512 + (s + kp) * 16 + el] = row[s];
    }
    for (int half = 0; half < 2; ++half) {
      int jh = j0 + half * 16;
      {
        int idx4 = tid * 4;
        *(v4f*)&v_s[idx4] =
            *(const v4f*)&v_g[((size_t)b * NN + jh + (idx4 >> 6)) * 64 + (idx4 & 63)];
      }
      __syncthreads();

      // layer 1: a1 = relu(u_i + v_j); packed 4-wide into A-fragment layout.
      {
        int p = tid, il = p >> 4, jl = p & 15;   // pair (i_local, j_local)
#pragma unroll
        for (int c = 0; c < 64; c += 4) {
          v4f uu = *(const v4f*)&u_s[il * 64 + c];
          v4f vv = *(const v4f*)&v_s[jl * 64 + c];
          v4h pk;
#pragma unroll
          for (int s = 0; s < 4; ++s) {
            float x = uu[s] + vv[s];
            pk[s] = (_Float16)(x > 0.f ? x : 0.f);
          }
          int q = c >> 5, k = c & 31;            // k%4 == 0
          int ln2 = jl + ((k >> 4) << 4);
          int el = k & 15;
          *(v4h*)&a1f[((il * 2 + q) * 32 + ln2) * 16 + el] = pk;
        }
      }
      __syncthreads();

      // layer 2: a2 = relu(a1 @ aw1 + ab1).  M-dim = j_local, tile index = i_local.
#pragma unroll
      for (int tt = 0; tt < 2; ++tt) {
        int tile = wave * 2 + tt;
        v8f acc[4];
#pragma unroll
        for (int ct = 0; ct < 4; ++ct) {
          v8f t;
#pragma unroll
          for (int r = 0; r < 8; ++r) t[r] = bias1[ct];
          acc[ct] = t;
        }
#pragma unroll
        for (int q = 0; q < 2; ++q) {
          v16h afr = *(const v16h*)&a1f[((tile * 2 + q) * 32 + lane) * 16];
#pragma unroll
          for (int ct = 0; ct < 4; ++ct)
            acc[ct] = __builtin_amdgcn_wmma_f32_16x16x32_f16(
                false, afr, false, w1r[ct][q], (short)0, acc[ct], false, false);
        }
        int Mb = (lane >= 16) ? 8 : 0;
        int nc = lane & 15;
#pragma unroll
        for (int ct = 0; ct < 4; ++ct)
#pragma unroll
          for (int r = 0; r < 8; ++r) {
            float x = acc[ct][r];
            x = x > 0.f ? x : 0.f;
            a2_s[(tile * 16 + r + Mb) * 64 + ct * 16 + nc] = (_Float16)x;
          }
      }
      __syncthreads();

      // layer 3: A = relu(a2 . aw2 + ab2).  a2 row read as 4x v16h (b128 LDS
      // loads); aw2 is wave-uniform -> scalar (SMEM) loads from global.
      {
        int p = tid, il = p >> 4, jl = p & 15;
        float acc = b2;
        const v16h* a2v = (const v16h*)&a2_s[p * 64];
#pragma unroll
        for (int cc = 0; cc < 4; ++cc) {
          v16h blk = a2v[cc];
#pragma unroll
          for (int s = 0; s < 16; ++s) acc += (float)blk[s] * aw2[cc * 16 + s];
        }
        acc = acc > 0.f ? acc : 0.f;
        int k = half * 16 + jl;                 // K index into 16x32 A fragment
        int ln2 = il + ((k >> 4) << 4);
        int el = k & 15;
        Af[ln2 * 16 + el] = (_Float16)acc;
      }
      __syncthreads();
    }

    // aggregation: Vacc += A(16x32) @ h(32x16), one WMMA per wave.
    {
      v16h afr = *(const v16h*)&Af[lane * 16];
      v16h hfr = *(const v16h*)&hf[(wave * 32 + lane) * 16];
      Vacc = __builtin_amdgcn_wmma_f32_16x16x32_f16(
          false, afr, false, hfr, (short)0, Vacc, false, false);
    }
    __syncthreads();
  }

  // write V per C/D layout
  {
    int Mb = (lane >= 16) ? 8 : 0;
    int nc = wave * 16 + (lane & 15);
#pragma unroll
    for (int r = 0; r < 8; ++r)
      V_g[((size_t)b * NN + i0 + r + Mb) * 128 + nc] = Vacc[r];
  }
}

// ---------------------------------------------------------------------------
// k3: update MLP + max-pool.  One block per 8 rows; non-negative float max
// via integer atomicMax (valid for relu outputs, out pre-zeroed by k0).
// ---------------------------------------------------------------------------
__global__ __launch_bounds__(256) void k3_update(
    const float* __restrict__ h32, const float* __restrict__ coords,
    const float* __restrict__ V_g,
    const float* __restrict__ uw0, const float* __restrict__ ub0,
    const float* __restrict__ uw1, const float* __restrict__ ub1,
    float* __restrict__ out)
{
  __shared__ float upd[8][260];
  __shared__ float t1[8][256];
  int r0 = blockIdx.x * 8;
  int t  = threadIdx.x;
  for (int e = t; e < 8 * 259; e += 256) {
    int rr = e / 259, k = e % 259, g = r0 + rr;
    float vv;
    if (k < 128)      vv = h32[(size_t)g * 128 + k];
    else if (k < 131) vv = coords[(size_t)g * 3 + (k - 128)];
    else              vv = V_g[(size_t)g * 128 + (k - 131)];
    upd[rr][k] = vv;
  }
  __syncthreads();
  {
    int c = t;
    float acc[8];
    float bb = ub0[c];
#pragma unroll
    for (int rr = 0; rr < 8; ++rr) acc[rr] = bb;
    for (int k = 0; k < 259; ++k) {
      float w = uw0[k * 256 + c];
#pragma unroll
      for (int rr = 0; rr < 8; ++rr) acc[rr] += upd[rr][k] * w;
    }
#pragma unroll
    for (int rr = 0; rr < 8; ++rr) t1[rr][c] = acc[rr] > 0.f ? acc[rr] : 0.f;
  }
  __syncthreads();
#pragma unroll
  for (int s = 0; s < 4; ++s) {
    int o = s * 256 + t;
    int rr = o >> 7, c = o & 127, g = r0 + rr;
    float acc = ub1[c];
#pragma unroll 8
    for (int k = 0; k < 256; ++k) acc += t1[rr][k] * uw1[k * 128 + c];
    acc = acc > 0.f ? acc : 0.f;
    int bat = g >> 9;                    // row / N
    atomicMax((int*)out + bat * 128 + c, __float_as_int(acc));
  }
}

// ---------------------------------------------------------------------------
extern "C" void kernel_launch(void* const* d_in, const int* in_sizes, int n_in,
                              void* d_out, int out_size, void* d_ws, size_t ws_size,
                              hipStream_t stream) {
  const float* coords = (const float*)d_in[0];
  const float* fw0 = (const float*)d_in[1];
  const float* fb0 = (const float*)d_in[2];
  const float* fw1 = (const float*)d_in[3];
  const float* fb1 = (const float*)d_in[4];
  const float* aw0 = (const float*)d_in[5];
  const float* ab0 = (const float*)d_in[6];
  const float* aw1 = (const float*)d_in[7];
  const float* ab1 = (const float*)d_in[8];
  const float* aw2 = (const float*)d_in[9];
  const float* ab2 = (const float*)d_in[10];
  const float* uw0 = (const float*)d_in[11];
  const float* ub0 = (const float*)d_in[12];
  const float* uw1 = (const float*)d_in[13];
  const float* ub1 = (const float*)d_in[14];

  char* ws = (char*)d_ws;
  float*    h32 = (float*)(ws + 0);                  // 8*512*128*4 = 2 MB
  float*    Vg  = (float*)(ws + (2u << 20));         // 2 MB
  float*    u_g = (float*)(ws + (4u << 20));         // 1 MB
  float*    v_g = (float*)(ws + (5u << 20));         // 1 MB
  _Float16* h16 = (_Float16*)(ws + (6u << 20));      // 1 MB

  k0_zero<<<(out_size + 255) / 256, 256, 0, stream>>>((float*)d_out, out_size);
  k1_node<<<BB * NN, 64, 0, stream>>>(coords, fw0, fb0, fw1, fb1, aw0,
                                      h32, h16, u_g, v_g);
  k2_adj_agg<<<BB * (NN / 16), 256, 0, stream>>>(u_g, v_g, h16, ab0, aw1, ab1,
                                                 aw2, ab2, Vg);
  k3_update<<<BB * NN / 8, 256, 0, stream>>>(h32, coords, Vg, uw0, ub0, uw1, ub1,
                                             (float*)d_out);
}
